// Contrast_loss_65120294142405
// MI455X (gfx1250) — compile-verified, matrix operand order checked by reading, not run on hardware
//
#include <hip/hip_runtime.h>
#include <hip/hip_bf16.h>
#include <math.h>

// ---------------------------------------------------------------------------
// Contrastive loss for MI455X (gfx1250, wave32).
// Compute-bound (~275 GFLOP vs ~35MB of input): use v_wmma_f32_16x16x32_bf16.
// Similarity matrices are reduced on the fly (exp row-sums) - never stored.
// Sim GEMM stages both operands in LDS (256KB of the 320KB WGP pool) to cut
// L2 traffic ~5x versus streaming fragments from global.
// ---------------------------------------------------------------------------

typedef __attribute__((ext_vector_type(16))) __bf16 v16bf;
typedef __attribute__((ext_vector_type(8)))  __bf16 v8bf;
typedef __attribute__((ext_vector_type(8)))  float  v8f;

#define INV_TAU 2.0f      // 1/0.5
#define NORM_EPS 1e-12f

// Load one 16x32 bf16 WMMA fragment for an NT GEMM (row-major operand, inner
// dim = K). Works for both A (rows of X) and B (rows of Y == cols of Y^T).
// Layout per ISA 7.12.2: lanes 0-15 hold K 0-7 (v0-3) and 16-23 (v4-7);
// lanes 16-31 hold K 8-15 and 24-31.
__device__ __forceinline__ v16bf load_frag_nt(const __bf16* base, int ld,
                                              int row0, int k0, int lane) {
    const int m = lane & 15;
    const int g = lane >> 4;
    const __bf16* p = base + (size_t)(row0 + m) * ld + (k0 + g * 8);
    v8bf lo = *(const v8bf*)(p);
    v8bf hi = *(const v8bf*)(p + 16);
    v16bf f;
#pragma unroll
    for (int i = 0; i < 8; ++i) { f[i] = lo[i]; f[i + 8] = hi[i]; }
    return f;
}

// ---------------------------------------------------------------------------
// Projection GEMM: C = A(MxK) * B(NxK)^T + bias, 128x128 block tile,
// 64x64 per wave (4x4 WMMA tiles). MODE 0: elu -> bf16. MODE 1: f32.
// Only ~6% of total FLOPs; operands stream straight from L2.
// ---------------------------------------------------------------------------
template <int MODE>
__global__ __launch_bounds__(128) void proj_gemm_nt(
    const unsigned short* Araw, const unsigned short* Braw, const float* bias,
    void* out, int N, int K) {
    const __bf16* A = (const __bf16*)Araw;
    const __bf16* B = (const __bf16*)Braw;
    const int lane = threadIdx.x & 31;
    const int wave = threadIdx.x >> 5;
    const int r0 = blockIdx.x * 128 + (wave >> 1) * 64;
    const int c0 = blockIdx.y * 128 + (wave & 1) * 64;

    v8f acc[4][4] = {};
    for (int k0 = 0; k0 < K; k0 += 32) {
        v16bf a[4], b[4];
#pragma unroll
        for (int t = 0; t < 4; ++t) a[t] = load_frag_nt(A, K, r0 + 16 * t, k0, lane);
#pragma unroll
        for (int t = 0; t < 4; ++t) b[t] = load_frag_nt(B, K, c0 + 16 * t, k0, lane);
#pragma unroll
        for (int tr = 0; tr < 4; ++tr)
#pragma unroll
            for (int tc = 0; tc < 4; ++tc)
                acc[tr][tc] = __builtin_amdgcn_wmma_f32_16x16x32_bf16(
                    false, a[tr], false, b[tc], (short)0, acc[tr][tc], false, false);
    }

    // C/D layout: vgpr r of lane l -> row r + 8*(l>>4), col l&15 (per 16x16 tile)
    const int g = lane >> 4, n = lane & 15;
#pragma unroll
    for (int tr = 0; tr < 4; ++tr)
#pragma unroll
        for (int tc = 0; tc < 4; ++tc)
#pragma unroll
            for (int r = 0; r < 8; ++r) {
                const int row = r0 + tr * 16 + g * 8 + r;
                const int col = c0 + tc * 16 + n;
                float v = acc[tr][tc][r] + bias[col];
                if (MODE == 0) {
                    v = (v > 0.0f) ? v : (__expf(v) - 1.0f);  // elu
                    ((__bf16*)out)[(size_t)row * N + col] = (__bf16)v;
                } else {
                    ((float*)out)[(size_t)row * N + col] = v;
                }
            }
}

// ---------------------------------------------------------------------------
// Similarity row-sum: rowsum[i] += sum_j exp(dot(x_i, y_j) / tau)
// Block = 4 waves in 2x2 -> 128x128 output tile per j-step; D fixed at 512.
// A slab (128 rows x 512, 128KB) staged once per block; B slab staged per
// j-step (128KB). Both slabs are contiguous in global since ld == K.
// gridDim.y slices the j-range; per-slice sums atomically accumulated.
// ---------------------------------------------------------------------------
__global__ __launch_bounds__(128) void sim_rowsum(
    const unsigned short* Xraw, const unsigned short* Yraw, float* rowsum,
    int jchunk) {
    constexpr int D = 512;
    __shared__ unsigned short shA[128 * D];   // 128 KB
    __shared__ unsigned short shB[128 * D];   // 128 KB

    const __bf16* X = (const __bf16*)Xraw;
    const __bf16* Y = (const __bf16*)Yraw;
    const int tid  = threadIdx.x;
    const int lane = tid & 31;
    const int wave = tid >> 5;
    const int rowbase = blockIdx.x * 128;
    const int j0 = blockIdx.y * jchunk;
    const int r0l = (wave >> 1) * 64;   // local row offset within 128-row tile
    const int c0l = (wave & 1) * 64;    // local col offset within 128-col tile

    // Stage A slab once (linear copy: rows are contiguous, ld == D).
    {
        const v8bf* gA = (const v8bf*)(X + (size_t)rowbase * D);
        v8bf* sA = (v8bf*)shA;
#pragma unroll
        for (int i = 0; i < (128 * D / 8) / 128; ++i)
            sA[tid + i * 128] = gA[tid + i * 128];
    }

    float rowacc[4][8] = {};

    for (int j = j0; j < j0 + jchunk; j += 128) {
        __syncthreads();  // previous j-step done reading shB (A copy on iter 0)
        {
            const v8bf* gB = (const v8bf*)(Y + (size_t)j * D);
            v8bf* sB = (v8bf*)shB;
#pragma unroll
            for (int i = 0; i < (128 * D / 8) / 128; ++i)
                sB[tid + i * 128] = gB[tid + i * 128];
        }
        __syncthreads();

        v8f acc[4][4] = {};
#pragma unroll 2
        for (int k0 = 0; k0 < D; k0 += 32) {
            v16bf a[4], b[4];
#pragma unroll
            for (int t = 0; t < 4; ++t)
                a[t] = load_frag_nt((const __bf16*)shA, D, r0l + 16 * t, k0, lane);
#pragma unroll
            for (int t = 0; t < 4; ++t)
                b[t] = load_frag_nt((const __bf16*)shB, D, c0l + 16 * t, k0, lane);
#pragma unroll
            for (int tr = 0; tr < 4; ++tr)
#pragma unroll
                for (int tc = 0; tc < 4; ++tc)
                    acc[tr][tc] = __builtin_amdgcn_wmma_f32_16x16x32_bf16(
                        false, a[tr], false, b[tc], (short)0, acc[tr][tc], false, false);
        }
#pragma unroll
        for (int tr = 0; tr < 4; ++tr)
#pragma unroll
            for (int r = 0; r < 8; ++r) {
                float s = 0.0f;
#pragma unroll
                for (int tc = 0; tc < 4; ++tc)
                    s += __expf(acc[tr][tc][r] * INV_TAU);
                rowacc[tr][r] += s;
            }
    }

    // Reduce across the 16 lanes that share a row (column dimension), then
    // lane n==0 of each half-wave commits its 32 rows.
    const int g = lane >> 4;
#pragma unroll
    for (int tr = 0; tr < 4; ++tr)
#pragma unroll
        for (int r = 0; r < 8; ++r) {
            float s = rowacc[tr][r];
            s += __shfl_xor(s, 1, 32);
            s += __shfl_xor(s, 2, 32);
            s += __shfl_xor(s, 4, 32);
            s += __shfl_xor(s, 8, 32);
            rowacc[tr][r] = s;
        }
    if ((lane & 15) == 0) {
#pragma unroll
        for (int tr = 0; tr < 4; ++tr)
#pragma unroll
            for (int r = 0; r < 8; ++r)
                atomicAdd(&rowsum[rowbase + r0l + tr * 16 + g * 8 + r], rowacc[tr][r]);
    }
}

// ---------------------------------------------------------------------------
// Elementwise f32 -> bf16 conversion.
// ---------------------------------------------------------------------------
__global__ void cvt_f32_to_bf16(const float* in, unsigned short* out, int n) {
    int i = blockIdx.x * blockDim.x + threadIdx.x;
    if (i < n) ((__bf16*)out)[i] = (__bf16)in[i];
}

// ---------------------------------------------------------------------------
// Row normalization (D = 512 fixed) + diagonal dot products (fp32 exact).
// ---------------------------------------------------------------------------
__device__ __forceinline__ float block_reduce_sum256(float v, float* sh) {
    const int t = threadIdx.x;
    sh[t] = v;
    __syncthreads();
    for (int s = 128; s > 0; s >>= 1) {
        if (t < s) sh[t] += sh[t + s];
        __syncthreads();
    }
    float r = sh[0];
    __syncthreads();
    return r;
}

__global__ __launch_bounds__(256) void normalize_diag(
    const float* H1, const float* H2, unsigned short* N1raw, unsigned short* N2raw,
    float* d11, float* d22, float* d12) {
    __shared__ float sh[256];
    const int row = blockIdx.x;
    const int t = threadIdx.x;
    const float* h1 = H1 + (size_t)row * 512;
    const float* h2 = H2 + (size_t)row * 512;
    const float a0 = h1[t], a1 = h1[t + 256];
    const float b0 = h2[t], b1 = h2[t + 256];
    const float ss1 = block_reduce_sum256(a0 * a0 + a1 * a1, sh);
    const float ss2 = block_reduce_sum256(b0 * b0 + b1 * b1, sh);
    const float inv1 = 1.0f / fmaxf(sqrtf(ss1), NORM_EPS);
    const float inv2 = 1.0f / fmaxf(sqrtf(ss2), NORM_EPS);
    const float n10 = a0 * inv1, n11 = a1 * inv1;
    const float n20 = b0 * inv2, n21 = b1 * inv2;
    __bf16* N1 = (__bf16*)N1raw;
    __bf16* N2 = (__bf16*)N2raw;
    N1[(size_t)row * 512 + t] = (__bf16)n10;
    N1[(size_t)row * 512 + t + 256] = (__bf16)n11;
    N2[(size_t)row * 512 + t] = (__bf16)n20;
    N2[(size_t)row * 512 + t + 256] = (__bf16)n21;
    const float dot = block_reduce_sum256(n10 * n20 + n11 * n21, sh);
    if (t == 0) {
        d11[row] = ss1 * inv1 * inv1;  // == sum(n1^2) ~ 1
        d22[row] = ss2 * inv2 * inv2;
        d12[row] = dot;
    }
}

// ---------------------------------------------------------------------------
// Final loss: mean over i of 0.5*(l1 + l2), single block.
// l1_i = log(rs11_i + rs12_i - exp(d11_i/tau)) - d12_i/tau
// l2_i = log(rs22_i + rs21_i - exp(d22_i/tau)) - d12_i/tau
// ---------------------------------------------------------------------------
__global__ __launch_bounds__(256) void loss_reduce(
    const float* rs11, const float* rs12, const float* rs22, const float* rs21,
    const float* d11, const float* d22, const float* d12, float* out, int Npts) {
    __shared__ float sh[256];
    float local = 0.0f;
    for (int i = threadIdx.x; i < Npts; i += 256) {
        const float num = d12[i] * INV_TAU;
        const float den1 = rs11[i] + rs12[i] - __expf(d11[i] * INV_TAU);
        const float den2 = rs22[i] + rs21[i] - __expf(d22[i] * INV_TAU);
        const float l1 = logf(den1) - num;
        const float l2 = logf(den2) - num;
        local += 0.5f * (l1 + l2);
    }
    const float total = block_reduce_sum256(local, sh);
    if (threadIdx.x == 0) out[0] = total / (float)Npts;
}

// ---------------------------------------------------------------------------
extern "C" void kernel_launch(void* const* d_in, const int* in_sizes, int n_in,
                              void* d_out, int out_size, void* d_ws, size_t ws_size,
                              hipStream_t stream) {
    const float* Z1 = (const float*)d_in[0];  // pri_embedding [8192,512]
    const float* Z2 = (const float*)d_in[1];  // aux_embedding [8192,512]
    const float* W1 = (const float*)d_in[2];  // [512,512]
    const float* b1 = (const float*)d_in[3];  // [512]
    const float* W2 = (const float*)d_in[4];  // [512,512]
    const float* b2 = (const float*)d_in[5];  // [512]
    float* out = (float*)d_out;

    constexpr int NPTS = 8192, D = 512;
    constexpr size_t SZ_WB = (size_t)D * D * 2;      // bf16 weights
    constexpr size_t SZ_ZB = (size_t)NPTS * D * 2;   // bf16 embeddings
    constexpr size_t SZ_HF = (size_t)NPTS * D * 4;   // f32 projections
    constexpr size_t SZ_V  = (size_t)NPTS * 4;       // per-row vectors

    char* w = (char*)d_ws;
    size_t off = 0;
    auto take = [&](size_t bytes) -> void* {
        void* p = w + off;
        off += (bytes + 255) & ~(size_t)255;
        return p;
    };
    unsigned short* W1bf = (unsigned short*)take(SZ_WB);
    unsigned short* W2bf = (unsigned short*)take(SZ_WB);
    unsigned short* Z1bf = (unsigned short*)take(SZ_ZB);
    unsigned short* Z2bf = (unsigned short*)take(SZ_ZB);
    unsigned short* M1bf = (unsigned short*)take(SZ_ZB);  // elu(Z1 W1^T + b1)
    unsigned short* M2bf = (unsigned short*)take(SZ_ZB);
    float* H1f = (float*)take(SZ_HF);
    float* H2f = (float*)take(SZ_HF);
    unsigned short* N1bf = (unsigned short*)take(SZ_ZB);  // normalized, bf16
    unsigned short* N2bf = (unsigned short*)take(SZ_ZB);
    float* rs11 = (float*)take(SZ_V);
    float* rs12 = (float*)take(SZ_V);
    float* rs22 = (float*)take(SZ_V);
    float* rs21 = (float*)take(SZ_V);
    float* d11  = (float*)take(SZ_V);
    float* d22  = (float*)take(SZ_V);
    float* d12  = (float*)take(SZ_V);

    // 1) convert weights + embeddings to bf16
    cvt_f32_to_bf16<<<(D * D + 255) / 256, 256, 0, stream>>>(W1, W1bf, D * D);
    cvt_f32_to_bf16<<<(D * D + 255) / 256, 256, 0, stream>>>(W2, W2bf, D * D);
    cvt_f32_to_bf16<<<(NPTS * D + 255) / 256, 256, 0, stream>>>(Z1, Z1bf, NPTS * D);
    cvt_f32_to_bf16<<<(NPTS * D + 255) / 256, 256, 0, stream>>>(Z2, Z2bf, NPTS * D);

    // 2) projection layer 1 (elu) and layer 2
    dim3 gProj(NPTS / 128, D / 128);
    proj_gemm_nt<0><<<gProj, 128, 0, stream>>>(Z1bf, W1bf, b1, M1bf, D, D);
    proj_gemm_nt<0><<<gProj, 128, 0, stream>>>(Z2bf, W1bf, b1, M2bf, D, D);
    proj_gemm_nt<1><<<gProj, 128, 0, stream>>>(M1bf, W2bf, b2, H1f, D, D);
    proj_gemm_nt<1><<<gProj, 128, 0, stream>>>(M2bf, W2bf, b2, H2f, D, D);

    // 3) normalize rows + diagonal dot products
    normalize_diag<<<NPTS, 256, 0, stream>>>(H1f, H2f, N1bf, N2bf, d11, d22, d12);

    // 4) similarity exp row-sums (4 directions), atomically accumulated
    hipMemsetAsync(rs11, 0, 4 * SZ_V, stream);  // rs11..rs21 are contiguous
    constexpr int JSLICES = 8;
    constexpr int JCHUNK = NPTS / JSLICES;  // 1024 cols -> 8 j-steps of 128
    dim3 gSim(NPTS / 128, JSLICES);
    sim_rowsum<<<gSim, 128, 0, stream>>>(N1bf, N1bf, rs11, JCHUNK);
    sim_rowsum<<<gSim, 128, 0, stream>>>(N1bf, N2bf, rs12, JCHUNK);
    sim_rowsum<<<gSim, 128, 0, stream>>>(N2bf, N2bf, rs22, JCHUNK);
    sim_rowsum<<<gSim, 128, 0, stream>>>(N2bf, N1bf, rs21, JCHUNK);

    // 5) final loss
    loss_reduce<<<1, 256, 0, stream>>>(rs11, rs12, rs22, rs21, d11, d22, d12, out, NPTS);
}
